// SMPLXFlowMatching_18021682774942
// MI455X (gfx1250) — matrix operand order repaired and checked by reading, not compiled, not run on hardware
//
#include <hip/hip_runtime.h>
#include <hip/hip_bf16.h>

// ---------------------------------------------------------------------------
// SMPLX flow-matching ODE sampler for gfx1250 (MI455X), bf16 WMMA pipeline.
// Round 2: templated GEMM with compile-time tile counts so accumulators stay
// in VGPRs (round 1 spilled acc[] to scratch due to dynamic indexing).
// ---------------------------------------------------------------------------

typedef __attribute__((ext_vector_type(16))) __bf16 v16bf;
typedef __attribute__((ext_vector_type(8)))  float  v8f;
typedef __attribute__((ext_vector_type(4)))  float  f32x4;

#define WG_THREADS 256
#define NWAVES     8
#define TILE_M     16
#define ACC_STRIDE 1028          // f32 scratch row stride (padded vs 1024)
#define HBF_STRIDE 1032          // bf16 activation row stride (16B-multiple)
#define APAD_CTX   2056          // bf16 stride for K=2048 ctx kernel
#define ZSTRIDE    144           // padded pose row (132 -> 144)
#define POSE_DIM   132
#define N_ROWS     8192
#define NUM_STEPS  75
#define LN_EPS     1e-5f

union Frag32B { v16bf v; f32x4 f4[2]; };

__device__ __forceinline__ float silu_f(float x) { return x / (1.f + __expf(-x)); }

// ---------------------------------------------------------------------------
// Workgroup GEMM: D(16 x 16*NTILES) = A(16 x 32*KSLABS, bf16 in LDS)
//                                   @ Bpacked(bf16, fragment-major in global)
// Waves split N: wave w owns tiles {w, w+8, ...} (NTILES % 8 == 0 required).
// A fragment, contiguous K order (same as pack_kernel B): lane L: M=L%16, half=L/16,
//   dwords 0-3 = K half*16..+7, dwords 4-7 = K half*16+8..+15 (two b128 loads).
// B fragment: 32 contiguous bytes per lane from the packed layout.
// Result -> lds_acc (f32, ACC_STRIDE row stride).
// ---------------------------------------------------------------------------
template<int KSLABS, int NTILES>
__device__ __forceinline__ void wg_gemm_bf16(
    const __hip_bfloat16* __restrict__ Bp,
    const __hip_bfloat16* lds_a, int a_stride,
    float* lds_acc)
{
  static_assert(NTILES % NWAVES == 0, "tiles must split evenly across waves");
  constexpr int TPW = NTILES / NWAVES;

  const int lane = threadIdx.x & 31;
  const int wave = threadIdx.x >> 5;
  const int half = lane >> 4;
  const int l16  = lane & 15;

  v8f acc[TPW];
#pragma unroll
  for (int i = 0; i < TPW; ++i)
#pragma unroll
    for (int r = 0; r < 8; ++r) acc[i][r] = 0.f;

  // Per-lane base into the packed-B stream for this wave's first tile.
  const __hip_bfloat16* bbase = Bp + ((size_t)wave * 32 + lane) * 16;

  for (int ks = 0; ks < KSLABS; ++ks) {
    Frag32B a;
    const __hip_bfloat16* ap = lds_a + l16 * a_stride + ks * 32 + half * 16;
    a.f4[0] = *(const f32x4*)(ap);        // K = base .. base+7
    a.f4[1] = *(const f32x4*)(ap + 8);    // K = base+8 .. base+15
#pragma unroll
    for (int i = 0; i < TPW; ++i) {
      Frag32B b;
      const __hip_bfloat16* bp =
          bbase + ((size_t)ks * NTILES + (size_t)i * NWAVES) * (32 * 16);
      b.f4[0] = *(const f32x4*)(bp);
      b.f4[1] = *(const f32x4*)(bp + 8);
      acc[i] = __builtin_amdgcn_wmma_f32_16x16x32_bf16(
          false, a.v, false, b.v, (short)0, acc[i], false, false);
    }
  }
  // D layout: lane L -> N = nt*16 + L%16, VGPR r -> M = r + 8*(L/16)
#pragma unroll
  for (int i = 0; i < TPW; ++i) {
    int nt = wave + i * NWAVES;
    float* dst = lds_acc + (half * 8) * ACC_STRIDE + nt * 16 + l16;
#pragma unroll
    for (int r = 0; r < 8; ++r) dst[r * ACC_STRIDE] = acc[i][r];
  }
}

// ---------------------------------------------------------------------------
// Weight packer: W(K x N, f32 row-major, leading dim ld, starting at rowOff)
// -> bf16 fragment-major layout expected by wg_gemm_bf16.  Zero-pads OOB.
// ---------------------------------------------------------------------------
__global__ void pack_kernel(const float* __restrict__ src, int rowOff, int ld,
                            int Kact, int Nact, int kslabs, int ntiles,
                            __hip_bfloat16* __restrict__ dst)
{
  int t = blockIdx.x * blockDim.x + threadIdx.x;
  int total = kslabs * ntiles * 32;
  if (t >= total) return;
  int lane = t & 31;
  int nt   = (t >> 5) % ntiles;
  int ks   = (t >> 5) / ntiles;
  int n     = nt * 16 + (lane & 15);
  int kbase = ks * 32 + (lane >> 4) * 16;
  __hip_bfloat16* o = dst + (size_t)t * 16;
  for (int i = 0; i < 16; ++i) {
    int k = kbase + i;
    float v = (k < Kact && n < Nact) ? src[(size_t)(rowOff + k) * ld + n] : 0.f;
    o[i] = __float2bfloat16(v);
  }
}

// ---------------------------------------------------------------------------
// t-embedding table: tpre[s][n] = sum_j silu(t_s*tw[j]+tb[j]) * iw[132+j][n]
// for s = 0..NUM_STEPS (76 distinct time points).  Tiny; plain VALU.
// ---------------------------------------------------------------------------
__global__ __launch_bounds__(256) void tpre_kernel(
    const float* __restrict__ tw, const float* __restrict__ tb,
    const float* __restrict__ iw, float* __restrict__ tpre)
{
  __shared__ float temb[256];
  const int s_ = blockIdx.x;
  const float t = (float)s_ * (1.f / NUM_STEPS);
  const int tid = threadIdx.x;
  temb[tid] = silu_f(t * tw[tid] + tb[tid]);
  __syncthreads();
  for (int n = tid; n < 1024; n += 256) {
    float acc = 0.f;
    for (int j = 0; j < 256; ++j)
      acc += temb[j] * iw[(size_t)(132 + j) * 1024 + n];
    tpre[(size_t)s_ * 1024 + n] = acc;
  }
}

// ---------------------------------------------------------------------------
// ctx precompute (per unique batch row, 1024 of them):
//   ctx_pre = (feats @ cw + cb) @ iw[388:1412] + ib       (16 rows per WG)
// ---------------------------------------------------------------------------
__global__ __launch_bounds__(256) void ctx_kernel(
    const float* __restrict__ feats, const float* __restrict__ cb,
    const float* __restrict__ ib,
    const __hip_bfloat16* __restrict__ wp_cw,
    const __hip_bfloat16* __restrict__ wp_iwctx,
    float* __restrict__ ctx_pre)
{
  __shared__ __align__(16) __hip_bfloat16 lds_a[TILE_M * APAD_CTX];
  __shared__ __align__(16) float          lds_acc[TILE_M * ACC_STRIDE];
  const int tid = threadIdx.x;
  const int base = blockIdx.x * TILE_M;

  for (int e = tid; e < TILE_M * 2048; e += WG_THREADS) {
    int r = e >> 11, c = e & 2047;
    lds_a[r * APAD_CTX + c] = __float2bfloat16(feats[(size_t)(base + r) * 2048 + c]);
  }
  __syncthreads();
  wg_gemm_bf16<64, 64>(wp_cw, lds_a, APAD_CTX, lds_acc);
  __syncthreads();
  for (int e = tid; e < TILE_M * 1024; e += WG_THREADS) {
    int r = e >> 10, c = e & 1023;
    lds_a[r * APAD_CTX + c] = __float2bfloat16(lds_acc[r * ACC_STRIDE + c] + cb[c]);
  }
  __syncthreads();
  wg_gemm_bf16<32, 64>(wp_iwctx, lds_a, APAD_CTX, lds_acc);
  __syncthreads();
  for (int e = tid; e < TILE_M * 1024; e += WG_THREADS) {
    int r = e >> 10, c = e & 1023;
    ctx_pre[(size_t)(base + r) * 1024 + c] = lds_acc[r * ACC_STRIDE + c] + ib[c];
  }
}

// ---------------------------------------------------------------------------
// Main ODE kernel: one WG owns 16 rows for all 75 Heun steps.
// ---------------------------------------------------------------------------
struct VFP {
  const __hip_bfloat16 *wp_iwx, *wp_w0, *wp_w1, *wp_ow;
  const float *tpre, *b0, *g0, *be0, *b1, *g1, *be1, *ob;
};

__device__ __forceinline__ void vector_field(
    const VFP& P, int t_idx,
    const float* zin, const float* lds_ctx,
    float* lds_acc, __hip_bfloat16* lds_hbf, float* lds_h1,
    float* vt_out, float* z_acc, const float* vt_prev, float halfdt)
{
  const int tid = threadIdx.x;

  // ---- pack x (K padded 132 -> 160) ----
  for (int e = tid; e < TILE_M * 160; e += WG_THREADS) {
    int r = e / 160, k = e - r * 160;
    float v = (k < POSE_DIM) ? zin[r * ZSTRIDE + k] : 0.f;
    lds_hbf[r * HBF_STRIDE + k] = __float2bfloat16(v);
  }
  __syncthreads();

  // ---- stage 1: x @ iw[0:132]  (K=160 padded, N=1024) ----
  wg_gemm_bf16<5, 64>(P.wp_iwx, lds_hbf, HBF_STRIDE, lds_acc);
  __syncthreads();
  const float* tp = P.tpre + (size_t)t_idx * 1024;
  for (int e = tid; e < TILE_M * 1024; e += WG_THREADS) {
    int r = e >> 10, c = e & 1023;
    float v = lds_acc[r * ACC_STRIDE + c] + lds_ctx[r * 1024 + c] + tp[c];
    float s = silu_f(v);
    lds_h1[r * 1024 + c] = s;
    lds_hbf[r * HBF_STRIDE + c] = __float2bfloat16(s);
  }
  __syncthreads();

  // ---- stage 2: h @ w0, LN, silu, residual ----
  wg_gemm_bf16<32, 64>(P.wp_w0, lds_hbf, HBF_STRIDE, lds_acc);
  __syncthreads();
  {
    const int lane = tid & 31, wave = tid >> 5;
    for (int rr = 0; rr < 2; ++rr) {
      int row = wave * 2 + rr;
      float s = 0.f, s2 = 0.f;
      for (int c = lane; c < 1024; c += 32) {
        float v = lds_acc[row * ACC_STRIDE + c] + P.b0[c];
        s += v; s2 += v * v;
      }
      for (int off = 16; off > 0; off >>= 1) {
        s  += __shfl_xor(s,  off, 32);
        s2 += __shfl_xor(s2, off, 32);
      }
      float mean = s * (1.f / 1024.f);
      float var  = s2 * (1.f / 1024.f) - mean * mean;
      float rstd = rsqrtf(var + LN_EPS);
      for (int c = lane; c < 1024; c += 32) {
        float v  = lds_acc[row * ACC_STRIDE + c] + P.b0[c];
        float ln = (v - mean) * rstd * P.g0[c] + P.be0[c];
        float h2 = lds_h1[row * 1024 + c] + silu_f(ln);
        lds_hbf[row * HBF_STRIDE + c] = __float2bfloat16(h2);
      }
    }
  }
  __syncthreads();

  // ---- stage 3: h @ w1 (N=512), LN, silu ----
  wg_gemm_bf16<32, 32>(P.wp_w1, lds_hbf, HBF_STRIDE, lds_acc);
  __syncthreads();
  {
    const int lane = tid & 31, wave = tid >> 5;
    for (int rr = 0; rr < 2; ++rr) {
      int row = wave * 2 + rr;
      float s = 0.f, s2 = 0.f;
      for (int c = lane; c < 512; c += 32) {
        float v = lds_acc[row * ACC_STRIDE + c] + P.b1[c];
        s += v; s2 += v * v;
      }
      for (int off = 16; off > 0; off >>= 1) {
        s  += __shfl_xor(s,  off, 32);
        s2 += __shfl_xor(s2, off, 32);
      }
      float mean = s * (1.f / 512.f);
      float var  = s2 * (1.f / 512.f) - mean * mean;
      float rstd = rsqrtf(var + LN_EPS);
      for (int c = lane; c < 512; c += 32) {
        float v  = lds_acc[row * ACC_STRIDE + c] + P.b1[c];
        float ln = (v - mean) * rstd * P.g1[c] + P.be1[c];
        lds_hbf[row * HBF_STRIDE + c] = __float2bfloat16(silu_f(ln));
      }
    }
  }
  __syncthreads();

  // ---- stage 4: h @ ow (K=512, N padded 132 -> 256, 16 tiles) ----
  wg_gemm_bf16<16, 16>(P.wp_ow, lds_hbf, HBF_STRIDE, lds_acc);
  __syncthreads();
  for (int e = tid; e < TILE_M * POSE_DIM; e += WG_THREADS) {
    int r = e / POSE_DIM, c = e - r * POSE_DIM;
    float v = lds_acc[r * ACC_STRIDE + c] + P.ob[c];
    if (vt_out) vt_out[r * ZSTRIDE + c] = v;                       // Heun 1st eval
    else        z_acc[r * ZSTRIDE + c] += halfdt * (vt_prev[r * ZSTRIDE + c] + v);
  }
  __syncthreads();
}

__global__ __launch_bounds__(256) void ode_kernel(
    const float* __restrict__ z_in, const float* __restrict__ ctx_pre,
    VFP P, float* __restrict__ out)
{
  __shared__ __align__(16) float          lds_ctx[TILE_M * 1024];
  __shared__ __align__(16) float          lds_acc[TILE_M * ACC_STRIDE];
  __shared__ __align__(16) float          lds_h1 [TILE_M * 1024];
  __shared__ __align__(16) __hip_bfloat16 lds_hbf[TILE_M * HBF_STRIDE];
  __shared__ __align__(16) float          lds_z  [TILE_M * ZSTRIDE];
  __shared__ __align__(16) float          lds_zp [TILE_M * ZSTRIDE];
  __shared__ __align__(16) float          lds_vt [TILE_M * ZSTRIDE];

  const int tid = threadIdx.x;
  const int rowBase = blockIdx.x * TILE_M;

  for (int e = tid; e < TILE_M * 1024; e += WG_THREADS) {
    int r = e >> 10, c = e & 1023;
    lds_ctx[e] = ctx_pre[(size_t)((rowBase + r) >> 3) * 1024 + c];   // repeat(feats,8)
  }
  for (int e = tid; e < TILE_M * POSE_DIM; e += WG_THREADS) {
    int r = e / POSE_DIM, c = e - r * POSE_DIM;
    lds_z[r * ZSTRIDE + c] = z_in[(size_t)(rowBase + r) * POSE_DIM + c];
  }
  __syncthreads();

  const float dt = 1.f / NUM_STEPS;
  for (int s_ = 0; s_ < NUM_STEPS; ++s_) {
    vector_field(P, s_,     lds_z,  lds_ctx, lds_acc, lds_hbf, lds_h1,
                 lds_vt, nullptr, nullptr, 0.f);
    for (int e = tid; e < TILE_M * POSE_DIM; e += WG_THREADS) {
      int r = e / POSE_DIM, c = e - r * POSE_DIM;
      lds_zp[r * ZSTRIDE + c] = lds_z[r * ZSTRIDE + c] + dt * lds_vt[r * ZSTRIDE + c];
    }
    __syncthreads();
    vector_field(P, s_ + 1, lds_zp, lds_ctx, lds_acc, lds_hbf, lds_h1,
                 nullptr, lds_z, lds_vt, 0.5f * dt);
  }

  // ---- outputs ----
  for (int e = tid; e < TILE_M * POSE_DIM; e += WG_THREADS) {
    int r = e / POSE_DIM, c = e - r * POSE_DIM;
    out[(size_t)(rowBase + r) * POSE_DIM + c] = lds_z[r * ZSTRIDE + c];
  }
  float* pose_out = out + (size_t)N_ROWS * POSE_DIM;
  for (int t = tid; t < TILE_M * 22; t += WG_THREADS) {
    int r = t / 22, j = t - r * 22;
    const float* p = lds_z + r * ZSTRIDE + j * 6;
    float a1x = p[0], a1y = p[1], a1z = p[2], a2x = p[3], a2y = p[4], a2z = p[5];
    float n1 = fmaxf(sqrtf(a1x*a1x + a1y*a1y + a1z*a1z), 1e-12f);
    float b1x = a1x/n1, b1y = a1y/n1, b1z = a1z/n1;
    float d = b1x*a2x + b1y*a2y + b1z*a2z;
    float c2x = a2x - d*b1x, c2y = a2y - d*b1y, c2z = a2z - d*b1z;
    float n2 = fmaxf(sqrtf(c2x*c2x + c2y*c2y + c2z*c2z), 1e-12f);
    float b2x = c2x/n2, b2y = c2y/n2, b2z = c2z/n2;
    float b3x = b1y*b2z - b1z*b2y, b3y = b1z*b2x - b1x*b2z, b3z = b1x*b2y - b1y*b2x;
    float* o = pose_out + ((size_t)(rowBase + r) * 22 + j) * 9;
    o[0] = b1x; o[1] = b2x; o[2] = b3x;     // stack axis=-1: columns b1,b2,b3
    o[3] = b1y; o[4] = b2y; o[5] = b3y;
    o[6] = b1z; o[7] = b2z; o[8] = b3z;
  }
  float* lp = out + (size_t)N_ROWS * POSE_DIM + (size_t)N_ROWS * 198;
  {
    const int lane = tid & 31, wave = tid >> 5;
    for (int rr = 0; rr < 2; ++rr) {
      int row = wave * 2 + rr;
      float s = 0.f;
      for (int c = lane; c < POSE_DIM; c += 32) {
        float v = lds_z[row * ZSTRIDE + c];
        s += v * v;
      }
      for (int off = 16; off > 0; off >>= 1) s += __shfl_xor(s, off, 32);
      if (lane == 0) lp[rowBase + row] = -0.5f * s / (float)POSE_DIM;
    }
  }
}

// ---------------------------------------------------------------------------
extern "C" void kernel_launch(void* const* d_in, const int* in_sizes, int n_in,
                              void* d_out, int out_size, void* d_ws, size_t ws_size,
                              hipStream_t stream)
{
  (void)in_sizes; (void)n_in; (void)out_size; (void)ws_size;
  const float* feats = (const float*)d_in[0];
  const float* z     = (const float*)d_in[1];
  const float* tw    = (const float*)d_in[2];
  const float* tb    = (const float*)d_in[3];
  const float* cw    = (const float*)d_in[4];
  const float* cb    = (const float*)d_in[5];
  const float* iw    = (const float*)d_in[6];
  const float* ib    = (const float*)d_in[7];
  const float* w0    = (const float*)d_in[8];
  const float* b0    = (const float*)d_in[9];
  const float* g0    = (const float*)d_in[10];
  const float* be0   = (const float*)d_in[11];
  const float* w1    = (const float*)d_in[12];
  const float* b1    = (const float*)d_in[13];
  const float* g1    = (const float*)d_in[14];
  const float* be1   = (const float*)d_in[15];
  const float* ow    = (const float*)d_in[16];
  const float* ob    = (const float*)d_in[17];

  char* ws = (char*)d_ws;
  size_t off = 0;
  auto carve = [&](size_t bytes) -> void* {
    void* p = ws + off;
    off = (off + bytes + 255) & ~(size_t)255;
    return p;
  };
  __hip_bfloat16* wp_cw    = (__hip_bfloat16*)carve((size_t)64 * 64 * 512 * 2);
  __hip_bfloat16* wp_iwctx = (__hip_bfloat16*)carve((size_t)32 * 64 * 512 * 2);
  __hip_bfloat16* wp_iwx   = (__hip_bfloat16*)carve((size_t) 5 * 64 * 512 * 2);
  __hip_bfloat16* wp_w0    = (__hip_bfloat16*)carve((size_t)32 * 64 * 512 * 2);
  __hip_bfloat16* wp_w1    = (__hip_bfloat16*)carve((size_t)32 * 32 * 512 * 2);
  __hip_bfloat16* wp_ow    = (__hip_bfloat16*)carve((size_t)16 * 16 * 512 * 2);
  float* tpre    = (float*)carve((size_t)(NUM_STEPS + 1) * 1024 * 4);
  float* ctx_pre = (float*)carve((size_t)1024 * 1024 * 4);

  auto pack = [&](const float* src, int ro, int ld, int K, int N,
                  int ks, int nt, __hip_bfloat16* dst) {
    int total = ks * nt * 32;
    pack_kernel<<<(total + 255) / 256, 256, 0, stream>>>(src, ro, ld, K, N, ks, nt, dst);
  };
  pack(cw,   0, 1024, 2048, 1024, 64, 64, wp_cw);
  pack(iw, 388, 1024, 1024, 1024, 32, 64, wp_iwctx);
  pack(iw,   0, 1024,  132, 1024,  5, 64, wp_iwx);
  pack(w0,   0, 1024, 1024, 1024, 32, 64, wp_w0);
  pack(w1,   0,  512, 1024,  512, 32, 32, wp_w1);
  pack(ow,   0,  132,  512,  132, 16, 16, wp_ow);

  tpre_kernel<<<NUM_STEPS + 1, 256, 0, stream>>>(tw, tb, iw, tpre);
  ctx_kernel<<<64, 256, 0, stream>>>(feats, cb, ib, wp_cw, wp_iwctx, ctx_pre);

  VFP P{wp_iwx, wp_w0, wp_w1, wp_ow, tpre, b0, g0, be0, b1, g1, be1, ob};
  ode_kernel<<<N_ROWS / TILE_M, 256, 0, stream>>>(z, ctx_pre, P, (float*)d_out);
}
